// HoulsbyAdapter_40355512714083
// MI455X (gfx1250) — matrix operand order repaired and checked by reading, not compiled, not run on hardware
//
#include <hip/hip_runtime.h>

#define D_MODEL 2048
#define ADAPTER 64
#define TOK_PER_BLK 16
#define THREADS 128

typedef __attribute__((ext_vector_type(16))) __bf16 v16bf;
typedef __attribute__((ext_vector_type(8)))  float  v8f;

union FragBF { uint4 u[2]; v16bf v; };

__device__ __forceinline__ unsigned short f2bf(float f) {
    // round-to-nearest-even fp32 -> bf16 (inputs here are finite)
    unsigned int u = __float_as_uint(f);
    u += 0x7FFFu + ((u >> 16) & 1u);
    return (unsigned short)(u >> 16);
}

// ---------------------------------------------------------------------------
// Prep: convert down_w (64x2048 f32) and up_w (2048x64 f32) into bf16,
// pre-swizzled into the WMMA B-fragment layout (ISA 7.12.2):
//   half i of lane holds element (K = kbase + i + 8*(i>=8) + 8*(lane>=16), N = lane%16)
// Bd: [ntile(4)][kstep(64)][lane(32)][half(16)]   (256 KB)
// Bu: [ntile(128)][kstep(2)][lane(32)][half(16)]  (256 KB)
// ---------------------------------------------------------------------------
__global__ void adapter_prep_weights(const float* __restrict__ down_w,
                                     const float* __restrict__ up_w,
                                     unsigned short* __restrict__ Bd,
                                     unsigned short* __restrict__ Bu) {
    int gid = blockIdx.x * blockDim.x + threadIdx.x;   // 262144 threads
    if (gid < 131072) {
        int i = gid & 15, lane = (gid >> 4) & 31, kstep = (gid >> 9) & 63, ntile = gid >> 15;
        int n = ntile * 16 + (lane & 15);
        int k = kstep * 32 + i + ((i >= 8) ? 8 : 0) + ((lane >= 16) ? 8 : 0);
        Bd[gid] = f2bf(down_w[n * D_MODEL + k]);
    } else {
        int g = gid - 131072;
        int i = g & 15, lane = (g >> 4) & 31, kstep = (g >> 9) & 1, ntile = g >> 10;
        int d = ntile * 16 + (lane & 15);
        int a = kstep * 32 + i + ((i >= 8) ? 8 : 0) + ((lane >= 16) ? 8 : 0);
        Bu[g] = f2bf(up_w[d * ADAPTER + a]);
    }
}

// ---------------------------------------------------------------------------
// Fused: LN -> down GEMM (WMMA bf16) -> relu -> up GEMM (WMMA bf16) -> residual
// One workgroup = 128 threads (4 waves) = 16 tokens.
// ---------------------------------------------------------------------------
__global__ __launch_bounds__(THREADS) void adapter_fused(
    const float* __restrict__ x,
    const float* __restrict__ ln_w, const float* __restrict__ ln_b,
    const unsigned short* __restrict__ Bd, const float* __restrict__ down_b,
    const unsigned short* __restrict__ Bu, const float* __restrict__ up_b,
    float* __restrict__ out) {

    __shared__ __align__(16) unsigned int   hU[TOK_PER_BLK * (D_MODEL / 2)]; // bf16 pairs, 64 KB
    __shared__ __align__(16) unsigned short h1s[TOK_PER_BLK * ADAPTER];      // 2 KB
    __shared__ float redS[16][8];
    __shared__ float redQ[16][8];
    __shared__ float muS[16];
    __shared__ float rsS[16];

    const int t      = threadIdx.x;
    const int token0 = blockIdx.x * TOK_PER_BLK;
    const int row    = t >> 3;   // 0..15 (token within tile)
    const int c8     = t & 7;    // 8 threads per token row

    const float4* x4   = (const float4*)x;
    const float4* lnw4 = (const float4*)ln_w;
    const float4* lnb4 = (const float4*)ln_b;

    // ---- Phase 1: LN statistics -------------------------------------------
    {
        const int rowg = token0 + row;
        float s = 0.f, q = 0.f;
        #pragma unroll 4
        for (int j = 0; j < 64; ++j) {
            float4 v = x4[rowg * (D_MODEL / 4) + c8 + j * 8];
            s += v.x + v.y + v.z + v.w;
            q += v.x * v.x + v.y * v.y + v.z * v.z + v.w * v.w;
        }
        redS[row][c8] = s;
        redQ[row][c8] = q;
    }
    __syncthreads();
    if (t < 16) {
        float s = 0.f, q = 0.f;
        #pragma unroll
        for (int j = 0; j < 8; ++j) { s += redS[t][j]; q += redQ[t][j]; }
        float mu  = s * (1.f / D_MODEL);
        float var = q * (1.f / D_MODEL) - mu * mu;
        muS[t] = mu;
        rsS[t] = rsqrtf(var + 1e-5f);
    }
    __syncthreads();

    // ---- Phase 2: normalize tile into LDS as bf16 (K-contiguous) ----------
    {
        const int   rowg = token0 + row;
        const float mu = muS[row], rs = rsS[row];
        for (int j = 0; j < 64; ++j) {
            int    i4 = c8 + j * 8;
            float4 v = x4[rowg * (D_MODEL / 4) + i4];
            float4 w = lnw4[i4];
            float4 b = lnb4[i4];
            float n0 = (v.x - mu) * rs * w.x + b.x;
            float n1 = (v.y - mu) * rs * w.y + b.y;
            float n2 = (v.z - mu) * rs * w.z + b.z;
            float n3 = (v.w - mu) * rs * w.w + b.w;
            hU[row * (D_MODEL / 2) + i4 * 2 + 0] = (unsigned)f2bf(n0) | ((unsigned)f2bf(n1) << 16);
            hU[row * (D_MODEL / 2) + i4 * 2 + 1] = (unsigned)f2bf(n2) | ((unsigned)f2bf(n3) << 16);
        }
    }
    __syncthreads();

    const int wave = t >> 5;       // 0..3
    const int lane = t & 31;
    const int lrow = lane & 15;    // M (A row) or N (B/C col)
    const int hi   = lane >> 4;    // lane group

    // ---- Phase 3: down GEMM 16x64x2048, relu, write h1 --------------------
    {
        v8f acc = {};
        const uint4* Bd4 = (const uint4*)Bd;
        for (int kstep = 0; kstep < 64; ++kstep) {
            const int kbase = kstep * 32;
            FragBF a, b;
            // A fragment (ISA 16-bit A layout): group0 halves = K[kbase..+7],[+16..+23]
            //                                   group1 halves = K[+8..+15],[+24..+31]
            const unsigned int* ap = &hU[lrow * (D_MODEL / 2) + (kbase >> 1) + hi * 4];
            a.u[0] = *(const uint4*)(ap);
            a.u[1] = *(const uint4*)(ap + 8);
            const uint4* bp = Bd4 + ((wave * 64 + kstep) * 32 + lane) * 2;
            b.u[0] = bp[0];
            b.u[1] = bp[1];
            acc = __builtin_amdgcn_wmma_f32_16x16x32_bf16(
                false, a.v, false, b.v, (short)0, acc, false, false);
        }
        const int   n  = wave * 16 + lrow;
        const float db = down_b[n];
        #pragma unroll
        for (int r = 0; r < 8; ++r) {
            int   m = r + hi * 8;              // C layout: VGPR r, lane group -> row
            float v = fmaxf(acc[r] + db, 0.f);
            h1s[m * ADAPTER + n] = f2bf(v);
        }
    }
    __syncthreads();

    // ---- Phase 4: up GEMM 16x2048x64 + bias + residual --------------------
    {
        FragBF a0, a1;  // K=0..31 and K=32..63 fragments, reused across 32 N-tiles
        {
            const unsigned short* p0 = &h1s[lrow * ADAPTER + 0  + hi * 8];
            const unsigned short* p1 = &h1s[lrow * ADAPTER + 32 + hi * 8];
            a0.u[0] = *(const uint4*)(p0);
            a0.u[1] = *(const uint4*)(p0 + 16);
            a1.u[0] = *(const uint4*)(p1);
            a1.u[1] = *(const uint4*)(p1 + 16);
        }
        const uint4* Bu4 = (const uint4*)Bu;
        for (int tt = 0; tt < 32; ++tt) {
            const int ntile = wave * 32 + tt;
            v8f acc = {};
            FragBF b0, b1;
            const uint4* bp = Bu4 + ((ntile * 2 + 0) * 32 + lane) * 2;
            b0.u[0] = bp[0]; b0.u[1] = bp[1];
            const uint4* bq = Bu4 + ((ntile * 2 + 1) * 32 + lane) * 2;
            b1.u[0] = bq[0]; b1.u[1] = bq[1];
            acc = __builtin_amdgcn_wmma_f32_16x16x32_bf16(
                false, a0.v, false, b0.v, (short)0, acc, false, false);
            acc = __builtin_amdgcn_wmma_f32_16x16x32_bf16(
                false, a1.v, false, b1.v, (short)0, acc, false, false);
            const int   d  = ntile * 16 + lrow;
            const float ub = up_b[d];
            #pragma unroll
            for (int r = 0; r < 8; ++r) {
                int tok = token0 + r + hi * 8;
                out[tok * D_MODEL + d] = acc[r] + ub + x[tok * D_MODEL + d];
            }
        }
    }
}

// ---------------------------------------------------------------------------
extern "C" void kernel_launch(void* const* d_in, const int* in_sizes, int n_in,
                              void* d_out, int out_size, void* d_ws, size_t ws_size,
                              hipStream_t stream) {
    const float* x      = (const float*)d_in[0];  // (4,4096,2048)
    const float* ln_w   = (const float*)d_in[1];  // (2048,)
    const float* ln_b   = (const float*)d_in[2];  // (2048,)
    const float* down_w = (const float*)d_in[3];  // (64,2048)
    const float* down_b = (const float*)d_in[4];  // (64,)
    const float* up_w   = (const float*)d_in[5];  // (2048,64)
    const float* up_b   = (const float*)d_in[6];  // (2048,)
    float*       out    = (float*)d_out;

    unsigned short* Bd = (unsigned short*)d_ws;   // 131072 halves (256 KB)
    unsigned short* Bu = Bd + 131072;             // 131072 halves (256 KB)

    // Pre-swizzle weights into WMMA B-fragment bf16 layout (L2-resident).
    adapter_prep_weights<<<1024, 256, 0, stream>>>(down_w, up_w, Bd, Bu);

    // 16384 tokens / 16 tokens-per-block = 1024 workgroups of 128 threads.
    adapter_fused<<<1024, THREADS, 0, stream>>>(x, ln_w, ln_b, Bd, down_b,
                                                Bu, up_b, out);
}
